// ManyBodyPadAttn_52295521796556
// MI455X (gfx1250) — matrix operand order repaired and checked by reading, not compiled, not run on hardware
//
#include <hip/hip_runtime.h>

// ---------------------------------------------------------------------------
// Types matching CDNA5 WMMA operand layouts (wave32)
// ---------------------------------------------------------------------------
typedef __bf16 bf16_t;
typedef __attribute__((ext_vector_type(16))) __bf16 v16bf;
typedef __attribute__((ext_vector_type(8)))  __bf16 v8bf;
typedef __attribute__((ext_vector_type(8)))  float  v8f;

__device__ inline v8f vzero8() {
    v8f v;
#pragma unroll
    for (int i = 0; i < 8; ++i) v[i] = 0.0f;
    return v;
}

__device__ inline v8f wmma_bf16(v16bf a, v16bf b, v8f c) {
    // D = A(16x32 bf16) * B(32x16 bf16) + C(16x16 f32)
    return __builtin_amdgcn_wmma_f32_16x16x32_bf16(
        /*neg_a=*/false, a, /*neg_b=*/false, b,
        /*c_mod=*/(short)0, c, /*reuse_a=*/false, /*reuse_b=*/false);
}

// Async copy 16B global -> LDS (per lane), tracked by ASYNCcnt (gfx1250).
__device__ inline void async_copy_b128(unsigned lds_off, const float* gptr) {
    asm volatile("global_load_async_to_lds_b128 %0, %1, off"
                 :: "v"(lds_off), "v"(gptr) : "memory");
}
__device__ inline void wait_asynccnt0() {
    asm volatile("s_wait_asynccnt 0x0" ::: "memory");
}

__device__ inline float half_max16(float v) {
#pragma unroll
    for (int m = 8; m >= 1; m >>= 1) v = fmaxf(v, __shfl_xor(v, m, 16));
    return v;
}
__device__ inline float half_sum16(float v) {
#pragma unroll
    for (int m = 8; m >= 1; m >>= 1) v += __shfl_xor(v, m, 16);
    return v;
}

#define BB 2
#define NN 128
#define CC 256
#define HEADS 8
#define HDIM 32
#define NEG_BIG (-3.4028234663852886e38f)

// ---------------------------------------------------------------------------
// Kernel 1: Q projection.  Q = (feat1 @ Wq^T + bq) * d^-0.5  -> bf16
// The 16x256 f32 activation strip is staged into LDS ONCE per block via
// GLOBAL_LOAD_ASYNC_TO_LDS_B128 (shared A-fragment for all 8 waves).
// Output layout: Qw[b][j][h][i][d]
// ---------------------------------------------------------------------------
__global__ __launch_bounds__(256) void proj_q_kernel(
    const float* __restrict__ feat1, const float* __restrict__ Wq,
    const float* __restrict__ bq, bf16_t* __restrict__ Qw) {
    __shared__ float stripF[16 * CC];  // 16 KB

    const int m0   = blockIdx.x * 16;
    const int lane = threadIdx.x & 31;
    const int wave = threadIdx.x >> 5;
    const int lh   = lane & 15;
    const bool hiH = lane >= 16;
    const int n0   = wave * 32;

    // ---- cooperative async stage: 1024 x b128 transfers, 4 per thread ----
    {
        const unsigned ldsbase = (unsigned)(uintptr_t)stripF;
        const float* gbase = feat1 + (size_t)m0 * CC;
#pragma unroll
        for (int it = 0; it < 4; ++it) {
            const int idx = threadIdx.x + it * 256;      // b128 index
            async_copy_b128(ldsbase + idx * 16, gbase + idx * 4);
        }
        wait_asynccnt0();
        __syncthreads();
    }

    const float* arow = stripF + lh * CC;  // LDS row for this lane
    const int ac0 = hiH ? 8 : 0;           // A-matrix K-chunk start
    const int bk0 = hiH ? 8 : 0;           // B-matrix K offset

    v8f acc[2];
    acc[0] = vzero8(); acc[1] = vzero8();

#pragma unroll
    for (int k0 = 0; k0 < CC; k0 += 32) {
        v8f f0 = *(const v8f*)(arow + k0 + ac0);
        v8f f1 = *(const v8f*)(arow + k0 + ac0 + 16);
        v16bf a;
#pragma unroll
        for (int i = 0; i < 8; ++i) { a[i] = (__bf16)f0[i]; a[i + 8] = (__bf16)f1[i]; }
#pragma unroll
        for (int j2 = 0; j2 < 2; ++j2) {
            const float* wrp = Wq + (size_t)(n0 + j2 * 16 + lh) * CC + k0 + bk0;
            __builtin_prefetch(wrp + 32, 0, 1);  // global_prefetch_b8
            v8f w0 = *(const v8f*)(wrp);
            v8f w1 = *(const v8f*)(wrp + 16);
            v16bf b;
#pragma unroll
            for (int i = 0; i < 8; ++i) { b[i] = (__bf16)w0[i]; b[i + 8] = (__bf16)w1[i]; }
            acc[j2] = wmma_bf16(a, b, acc[j2]);
        }
    }

    const float scale = 0.17677669529663687f;  // 32^-0.5
#pragma unroll
    for (int j2 = 0; j2 < 2; ++j2) {
        const int ch = n0 + j2 * 16 + lh;
        const float bias = bq[ch];
        const int h = ch & 7, d = ch >> 3;
#pragma unroll
        for (int r = 0; r < 8; ++r) {
            const int M = hiH ? r + 8 : r;
            const int m = m0 + M;
            const int b  = m >> 14;
            const int i  = (m >> 7) & 127;
            const int jj = m & 127;
            const float v = (acc[j2][r] + bias) * scale;
            Qw[((size_t)((b * NN + jj) * HEADS + h) * NN + i) * HDIM + d] = (__bf16)v;
        }
    }
}

// ---------------------------------------------------------------------------
// Kernel 2: KV projection. KV = feat2 @ Wkv^T + bkv  -> bf16
//   ch < 256 -> K:  Kw[b][j][h][k][d]
//   ch >=256 -> V stored TRANSPOSED: Vw[b][j][h][d][k]
// Same async-LDS staging of the shared activation strip.
// ---------------------------------------------------------------------------
__global__ __launch_bounds__(256) void proj_kv_kernel(
    const float* __restrict__ feat2, const float* __restrict__ Wkv,
    const float* __restrict__ bkv, bf16_t* __restrict__ Kw,
    bf16_t* __restrict__ Vw) {
    __shared__ float stripF[16 * CC];  // 16 KB

    const int m0   = blockIdx.x * 16;
    const int lane = threadIdx.x & 31;
    const int wave = threadIdx.x >> 5;
    const int lh   = lane & 15;
    const bool hiH = lane >= 16;
    const int n0   = wave * 64;

    {
        const unsigned ldsbase = (unsigned)(uintptr_t)stripF;
        const float* gbase = feat2 + (size_t)m0 * CC;
#pragma unroll
        for (int it = 0; it < 4; ++it) {
            const int idx = threadIdx.x + it * 256;
            async_copy_b128(ldsbase + idx * 16, gbase + idx * 4);
        }
        wait_asynccnt0();
        __syncthreads();
    }

    const float* arow = stripF + lh * CC;
    const int ac0 = hiH ? 8 : 0;
    const int bk0 = hiH ? 8 : 0;

    v8f acc[4];
#pragma unroll
    for (int t = 0; t < 4; ++t) acc[t] = vzero8();

#pragma unroll
    for (int k0 = 0; k0 < CC; k0 += 32) {
        v8f f0 = *(const v8f*)(arow + k0 + ac0);
        v8f f1 = *(const v8f*)(arow + k0 + ac0 + 16);
        v16bf a;
#pragma unroll
        for (int i = 0; i < 8; ++i) { a[i] = (__bf16)f0[i]; a[i + 8] = (__bf16)f1[i]; }
#pragma unroll
        for (int j2 = 0; j2 < 4; ++j2) {
            const float* wrp = Wkv + (size_t)(n0 + j2 * 16 + lh) * CC + k0 + bk0;
            v8f w0 = *(const v8f*)(wrp);
            v8f w1 = *(const v8f*)(wrp + 16);
            v16bf b;
#pragma unroll
            for (int i = 0; i < 8; ++i) { b[i] = (__bf16)w0[i]; b[i + 8] = (__bf16)w1[i]; }
            acc[j2] = wmma_bf16(a, b, acc[j2]);
        }
    }

#pragma unroll
    for (int j2 = 0; j2 < 4; ++j2) {
        const int ch = n0 + j2 * 16 + lh;       // 0..511
        const float bias = bkv[ch];
#pragma unroll
        for (int r = 0; r < 8; ++r) {
            const int M = hiH ? r + 8 : r;
            const int m = m0 + M;
            const int b = m >> 14;
            const int j = (m >> 7) & 127;
            const int k = m & 127;
            const float v = acc[j2][r] + bias;
            if (ch < CC) {
                const int h = ch & 7, d = ch >> 3;
                Kw[((size_t)((b * NN + j) * HEADS + h) * NN + k) * HDIM + d] = (__bf16)v;
            } else {
                const int c2 = ch - CC;
                const int h = c2 & 7, d = c2 >> 3;
                Vw[((size_t)((b * NN + j) * HEADS + h) * HDIM + d) * NN + k] = (__bf16)v;
            }
        }
    }
}

// ---------------------------------------------------------------------------
// Kernel 3: EG projection (tiny N=16) + sigmoid(G).  f32 outputs.
// Ew[b][j][h][k], Gw[b][j][h][k] (= sigmoid already applied)
// ---------------------------------------------------------------------------
__global__ __launch_bounds__(256) void proj_eg_kernel(
    const float* __restrict__ feat2, const float* __restrict__ Weg,
    const float* __restrict__ beg, float* __restrict__ Ew,
    float* __restrict__ Gw) {
    const int gid = blockIdx.x * blockDim.x + threadIdx.x;  // 0 .. 262143
    const int m = gid >> 3;
    const int h = gid & 7;
    const float* x  = feat2 + (size_t)m * CC;
    const float* we = Weg + (size_t)h * CC;
    const float* wg = Weg + (size_t)(h + HEADS) * CC;
    float e = 0.f, g = 0.f;
#pragma unroll 8
    for (int c = 0; c < CC; ++c) {
        const float xv = x[c];
        e = fmaf(xv, we[c], e);
        g = fmaf(xv, wg[c], g);
    }
    e += beg[h];
    g += beg[h + HEADS];
    const float gs = 1.0f / (1.0f + __expf(-g));
    const int b = m >> 14, j = (m >> 7) & 127, k = m & 127;
    const size_t idx = ((size_t)(b * NN + j) * HEADS + h) * NN + k;
    Ew[idx] = e;
    Gw[idx] = gs;
}

// ---------------------------------------------------------------------------
// Kernel 4: attention core.  1 block per (b,j); wave w handles head h=w.
//   S = Q K^T + E, mask, softmax_k, * sigmoid(G); Va = A V.
// K/V fragments pinned in VGPRs; A staged through LDS in bf16.
// ---------------------------------------------------------------------------
__global__ __launch_bounds__(256) void attn_kernel(
    const bf16_t* __restrict__ Qw, const bf16_t* __restrict__ Kw,
    const bf16_t* __restrict__ Vw, const float* __restrict__ Ew,
    const float* __restrict__ Gw, const unsigned char* __restrict__ mask,
    float* __restrict__ Vaw) {
    __shared__ __bf16 As[HEADS][16 * NN];  // 32 KB

    const int bj = blockIdx.x;
    const int b = bj >> 7, j = bj & 127;
    const int lane = threadIdx.x & 31;
    const int h = threadIdx.x >> 5;
    const int lh = lane & 15;
    const bool hiH = lane >= 16;
    const int koff = hiH ? 16 : 0;
    const int c0 = hiH ? 16 : 0;

    const size_t base = (size_t)bj * HEADS + h;
    const bf16_t* qb = Qw + base * (NN * HDIM);
    const bf16_t* kp = Kw + base * (NN * HDIM);
    const bf16_t* vp = Vw + base * (HDIM * NN);
    const float* eb = Ew + base * NN;
    const float* gb = Gw + base * NN;

    // Persistent B fragments: K^T (8 column tiles) and V (4 k-chunks x 2 d-halves)
    v16bf kf[8];
#pragma unroll
    for (int t = 0; t < 8; ++t)
        kf[t] = *(const v16bf*)(kp + (size_t)(t * 16 + lh) * HDIM + koff);
    v16bf vf[8];
#pragma unroll
    for (int c = 0; c < 4; ++c)
#pragma unroll
        for (int e = 0; e < 2; ++e)
            vf[2 * c + e] = *(const v16bf*)(vp + (size_t)(e * 16 + lh) * NN + c * 32 + koff);

    float er[8], gr[8];
#pragma unroll
    for (int t = 0; t < 8; ++t) { er[t] = eb[t * 16 + lh]; gr[t] = gb[t * 16 + lh]; }

    for (int si = 0; si < 8; ++si) {
        // ---- A fragment: Q strip rows [si*16 .. si*16+15], K-dim = 32 ----
        const bf16_t* qp = qb + (size_t)(si * 16 + lh) * HDIM + c0;
        v8bf q0 = *(const v8bf*)qp;
        v8bf q1 = *(const v8bf*)(qp + 8);
        v16bf qa;
#pragma unroll
        for (int i = 0; i < 8; ++i) { qa[i] = q0[i]; qa[i + 8] = q1[i]; }

        // ---- S = Q K^T (one WMMA per 16x16 tile, k-dim 32 exactly) ----
        v8f acc[8];
#pragma unroll
        for (int t = 0; t < 8; ++t) acc[t] = wmma_bf16(qa, kf[t], vzero8());

        // ---- + E, apply mask ----
#pragma unroll
        for (int t = 0; t < 8; ++t) {
            const int k = t * 16 + lh;
#pragma unroll
            for (int r = 0; r < 8; ++r) {
                const int i = si * 16 + (hiH ? r + 8 : r);
                const unsigned char mv =
                    mask[(((size_t)(b * NN + i) * NN + j) * NN) + k];
                const float v = acc[t][r] + er[t];
                acc[t][r] = mv ? v : NEG_BIG;
            }
        }

        // ---- row softmax over k (row's 16 cols live in one 16-lane half) ----
#pragma unroll
        for (int r = 0; r < 8; ++r) {
            float mx = acc[0][r];
#pragma unroll
            for (int t = 1; t < 8; ++t) mx = fmaxf(mx, acc[t][r]);
            mx = half_max16(mx);
            float sum = 0.f;
#pragma unroll
            for (int t = 0; t < 8; ++t) {
                const float e = __expf(acc[t][r] - mx);
                acc[t][r] = e;
                sum += e;
            }
            sum = half_sum16(sum);
            const float inv = 1.0f / sum;
#pragma unroll
            for (int t = 0; t < 8; ++t) acc[t][r] *= inv * gr[t];
        }

        // ---- stage probabilities to LDS as bf16 (row-major 16x128) ----
#pragma unroll
        for (int t = 0; t < 8; ++t) {
            const int k = t * 16 + lh;
#pragma unroll
            for (int r = 0; r < 8; ++r) {
                const int M = hiH ? r + 8 : r;
                As[h][M * NN + k] = (__bf16)acc[t][r];
            }
        }
        // LDS ops from one wave are in-order (DScnt): store->load is safe.

        // ---- Va strip = A (16x128) * V (128x32): 4 k-chunks x 2 d-tiles ----
        v8f o0 = vzero8(), o1 = vzero8();
#pragma unroll
        for (int c = 0; c < 4; ++c) {
            const __bf16* ap = &As[h][lh * NN + c * 32 + c0];
            v8bf a0 = *(const v8bf*)ap;
            v8bf a1 = *(const v8bf*)(ap + 8);
            v16bf af;
#pragma unroll
            for (int i = 0; i < 8; ++i) { af[i] = a0[i]; af[i + 8] = a1[i]; }
            o0 = wmma_bf16(af, vf[2 * c + 0], o0);
            o1 = wmma_bf16(af, vf[2 * c + 1], o1);
        }

        // ---- scatter Va (pre-LN) as f32: channel ch = d*8 + h ----
#pragma unroll
        for (int e = 0; e < 2; ++e) {
            const int d = e * 16 + lh;
            const int ch = d * HEADS + h;
#pragma unroll
            for (int r = 0; r < 8; ++r) {
                const int i = si * 16 + (hiH ? r + 8 : r);
                Vaw[((size_t)(b * NN + i) * NN + j) * CC + ch] =
                    (e == 0 ? o0[r] : o1[r]);
            }
        }
    }
}

// ---------------------------------------------------------------------------
// Kernel 5: LayerNorm over C=256.  Wave per row, 8 channels per lane.
// ---------------------------------------------------------------------------
__global__ __launch_bounds__(256) void ln_kernel(
    const float* __restrict__ Vaw, const float* __restrict__ gamma,
    const float* __restrict__ beta, float* __restrict__ out) {
    const int lane = threadIdx.x & 31;
    const int wave = threadIdx.x >> 5;
    const int row = blockIdx.x * 8 + wave;
    const float* x = Vaw + (size_t)row * CC;
    float v[8];
    float s = 0.f, s2 = 0.f;
#pragma unroll
    for (int i = 0; i < 8; ++i) {
        const float t = x[lane * 8 + i];
        v[i] = t;
        s += t;
        s2 = fmaf(t, t, s2);
    }
#pragma unroll
    for (int m = 16; m >= 1; m >>= 1) {
        s += __shfl_xor(s, m, 32);
        s2 += __shfl_xor(s2, m, 32);
    }
    const float mean = s * (1.0f / CC);
    const float var = s2 * (1.0f / CC) - mean * mean;
    const float inv = rsqrtf(var + 1e-3f);
    float* o = out + (size_t)row * CC;
#pragma unroll
    for (int i = 0; i < 8; ++i) {
        const int ch = lane * 8 + i;
        o[ch] = (v[i] - mean) * inv * gamma[ch] + beta[ch];
    }
}

// ---------------------------------------------------------------------------
// Host launcher
// ---------------------------------------------------------------------------
extern "C" void kernel_launch(void* const* d_in, const int* in_sizes, int n_in,
                              void* d_out, int out_size, void* d_ws, size_t ws_size,
                              hipStream_t stream) {
    (void)in_sizes; (void)n_in; (void)out_size; (void)ws_size;
    const float* feat1 = (const float*)d_in[0];
    const float* feat2 = (const float*)d_in[1];
    const unsigned char* mask = (const unsigned char*)d_in[2];
    const float* Wq   = (const float*)d_in[3];
    const float* bq   = (const float*)d_in[4];
    const float* Wkv  = (const float*)d_in[5];
    const float* bkv  = (const float*)d_in[6];
    const float* Weg  = (const float*)d_in[7];
    const float* beg  = (const float*)d_in[8];
    const float* ln_g = (const float*)d_in[9];
    const float* ln_b = (const float*)d_in[10];
    float* out = (float*)d_out;

    // Workspace carve-up (bytes)
    char* ws = (char*)d_ws;
    const size_t headElems = (size_t)BB * NN * HEADS * NN * HDIM;  // 8,388,608
    bf16_t* Qw = (bf16_t*)(ws);                                    // 16 MB
    bf16_t* Kw = (bf16_t*)(ws + headElems * 2);                    // 16 MB
    bf16_t* Vw = (bf16_t*)(ws + headElems * 4);                    // 16 MB
    float*  Ew = (float*)(ws + headElems * 6);                     // 1 MB
    float*  Gw = (float*)(ws + headElems * 6 + (size_t)BB * NN * HEADS * NN * 4);
    float*  Vaw = (float*)(ws + headElems * 6 + (size_t)2 * BB * NN * HEADS * NN * 4);

    const int Mrows = BB * NN * NN;  // 32768

    proj_q_kernel<<<dim3(Mrows / 16), dim3(256), 0, stream>>>(feat1, Wq, bq, Qw);
    proj_kv_kernel<<<dim3(Mrows / 16), dim3(256), 0, stream>>>(feat2, Wkv, bkv, Kw, Vw);
    proj_eg_kernel<<<dim3((Mrows * HEADS) / 256), dim3(256), 0, stream>>>(feat2, Weg, beg, Ew, Gw);
    attn_kernel<<<dim3(BB * NN), dim3(256), 0, stream>>>(Qw, Kw, Vw, Ew, Gw, mask, Vaw);
    ln_kernel<<<dim3(Mrows / 8), dim3(256), 0, stream>>>(Vaw, ln_g, ln_b, out);
}
